// DualTransformerCritic_77558519431826
// MI455X (gfx1250) — compile-verified, matrix operand order checked by reading, not run on hardware
//
#include <hip/hip_runtime.h>
#include <hip/hip_bf16.h>
#include <cmath>

typedef __attribute__((ext_vector_type(16))) _Float16 v16h;
typedef __attribute__((ext_vector_type(8)))  _Float16 v8h;
typedef __attribute__((ext_vector_type(8)))  float    v8f;

#define NB   2048
#define NAG  64      // agents
#define DOBS 256
#define NACT 64
#define DH   128
#define DOA  320
#define INV_SQRT_DK 0.08838834764831845f

// workspace offsets (in halves) for f16 weights
#define OFF_WPRE 0
#define OFF_WK   32768
#define OFF_WQ   49152
#define OFF_WSE  65536
#define OFF_WSA  81920
#define OFF_WAV  122880
#define OFF_WSQ  139264
#define OFF_WF1  155648

#define SMEM_BYTES 147456

// ---------------- weight fp32 -> f16 conversion ----------------
__global__ void cvt_f32_f16(const float* __restrict__ s, _Float16* __restrict__ d, int n) {
  int i = blockIdx.x * blockDim.x + threadIdx.x;
  if (i < n) d[i] = (_Float16)s[i];
}

// ---------------- WMMA fragment helpers ----------------
// Loads one 16x32 operand fragment (A layout; identical per-lane pattern used
// for B by passing the B^T row-major array). Per lane: two contiguous 16-byte
// chunks: rows r0+(lane&15), K in [k0+kb, k0+kb+7] and [k0+kb+16, k0+kb+23],
// kb = (lane>>4)*8.  Requires ld % 8 == 0 and k0 % 32 == 0 (always true here).
__device__ __forceinline__ v16h load_frag(const _Float16* __restrict__ p0,
                                          int ld, int r0, int k0, int lane) {
  const _Float16* p = p0 + (r0 + (lane & 15)) * ld + k0 + ((lane >> 4) << 3);
  v8h lo = *reinterpret_cast<const v8h*>(p);
  v8h hi = *reinterpret_cast<const v8h*>(p + 16);
  v16h r;
#pragma unroll
  for (int i = 0; i < 8; ++i) { r[i] = lo[i]; r[8 + i] = hi[i]; }
  return r;
}

// C[64,N] = act( A[64,K] @ Bt[N,K]^T + bias ), f16 out to LDS (+ opt transpose).
// Each wave owns one 16-wide N-strip: all K/32 B-fragments for the strip are
// loaded from (global) memory ONCE into registers, then the wave sweeps its 4 (N=128) or 2 (N=64)
// M-tiles reading A from LDS. Weight traffic = 1x per workgroup for N=128, 2x for N=64.
template <int N, int K, bool ACT, bool BIAS, bool TRN>
__device__ __forceinline__ void gemm_f16(const _Float16* __restrict__ A, int lda,
                                         const _Float16* __restrict__ Bt, int ldb,
                                         _Float16* __restrict__ Out, int ldo,
                                         _Float16* __restrict__ OutT, int ldt,
                                         const float* __restrict__ bias, int tid) {
  constexpr int KF = K / 32;
  const int wid = tid >> 5, lane = tid & 31;
  int n0, mtLo, mtN;
  if (N == 128) { n0 = wid << 4;        mtLo = 0;               mtN = 4; }
  else          { n0 = (wid & 3) << 4;  mtLo = (wid >> 2) << 1; mtN = 2; }

  v16h bf[KF];
#pragma unroll
  for (int ki = 0; ki < KF; ++ki) bf[ki] = load_frag(Bt, ldb, n0, ki * 32, lane);

  const int n  = n0 + (lane & 15);
  const float bv = BIAS ? bias[n] : 0.0f;

  for (int mi = 0; mi < mtN; ++mi) {
    const int m0 = (mtLo + mi) << 4;
    v8f c = {};
#pragma unroll
    for (int ki = 0; ki < KF; ++ki) {
      v16h a = load_frag(A, lda, m0, ki * 32, lane);
      c = __builtin_amdgcn_wmma_f32_16x16x32_f16(false, a, false, bf[ki],
                                                 (short)0, c, false, false);
    }
    const int mb = m0 + ((lane >> 4) << 3);
#pragma unroll
    for (int v = 0; v < 8; ++v) {
      float x = c[v] + bv;
      if (ACT) x = (x > 0.0f) ? x : 0.01f * x;   // leaky_relu(0.01)
      Out[(mb + v) * ldo + n] = (_Float16)x;
      if (TRN) OutT[n * ldt + (mb + v)] = (_Float16)x;
    }
  }
}

// scores[64,64] = scale * (A[64,K] @ Bt[64,K]^T), fp32 to LDS
template <int K>
__device__ __forceinline__ void gemm_f32(const _Float16* __restrict__ A, int lda,
                                         const _Float16* __restrict__ Bt, int ldb,
                                         float* __restrict__ Out, float scale, int tid) {
  constexpr int KF = K / 32;
  const int wid = tid >> 5, lane = tid & 31;
  const int n0   = (wid & 3) << 4;
  const int mtLo = (wid >> 2) << 1;

  v16h bf[KF];
#pragma unroll
  for (int ki = 0; ki < KF; ++ki) bf[ki] = load_frag(Bt, ldb, n0, ki * 32, lane);

  const int n = n0 + (lane & 15);
  for (int mi = 0; mi < 2; ++mi) {
    const int m0 = (mtLo + mi) << 4;
    v8f c = {};
#pragma unroll
    for (int ki = 0; ki < KF; ++ki) {
      v16h a = load_frag(A, lda, m0, ki * 32, lane);
      c = __builtin_amdgcn_wmma_f32_16x16x32_f16(false, a, false, bf[ki],
                                                 (short)0, c, false, false);
    }
    const int mb = m0 + ((lane >> 4) << 3);
#pragma unroll
    for (int v = 0; v < 8; ++v) Out[(mb + v) * 64 + n] = c[v] * scale;
  }
}

// row softmax over [64,64] fp32 scores: 4 lanes per row, shfl_xor reduce.
// Writes normalized probs as f16 to LDS and fp32 to global.
__device__ __forceinline__ void softmax64(float* __restrict__ S,
                                          _Float16* __restrict__ P,
                                          float* __restrict__ gOut, int tid) {
  const int r = tid >> 2, q = tid & 3;
  float* row = S + r * 64;
  float m = -3.0e38f;
#pragma unroll
  for (int j = 0; j < 16; ++j) m = fmaxf(m, row[q * 16 + j]);
  m = fmaxf(m, __shfl_xor(m, 1, 32));
  m = fmaxf(m, __shfl_xor(m, 2, 32));
  float s = 0.0f;
#pragma unroll
  for (int j = 0; j < 16; ++j) {
    float e = __expf(row[q * 16 + j] - m);
    row[q * 16 + j] = e;
    s += e;
  }
  s += __shfl_xor(s, 1, 32);
  s += __shfl_xor(s, 2, 32);
  const float inv = 1.0f / s;
#pragma unroll
  for (int j = 0; j < 16; ++j) {
    float p = row[q * 16 + j] * inv;
    P[r * 64 + q * 16 + j]    = (_Float16)p;
    gOut[r * 64 + q * 16 + j] = p;
  }
}

// ---------------- fully fused critic: one workgroup per batch item ----------------
__global__ void __launch_bounds__(256)
critic_fused(const float* __restrict__ states, const float* __restrict__ actions,
             const float* __restrict__ b_pre, const float* __restrict__ b_se,
             const float* __restrict__ b_sa,  const float* __restrict__ b_sq,
             const float* __restrict__ W_f2,  const _Float16* __restrict__ ws,
             float* __restrict__ out_value, float* __restrict__ out_w1,
             float* __restrict__ out_w2) {
  extern __shared__ __align__(16) char smem[];
  // LDS plan (144 KB), with phase-based reuse:
  _Float16* oa     = (_Float16*)(smem);            // [64,320] obs||act    (later av/avT)
  _Float16* av     = (_Float16*)(smem);            // [64,128]
  _Float16* avT    = (_Float16*)(smem + 16384);    // [128,64]
  _Float16* sbufA  = (_Float16*)(smem + 40960);    // [64,128] se_pre, then se
  _Float16* sbufB  = (_Float16*)(smem + 57344);    // [128,64] se_preT, then oae[64,128]
  _Float16* qbuf   = (_Float16*)(smem + 73728);    // [64,128]   } together: nf [64,256]
  _Float16* kbuf   = (_Float16*)(smem + 90112);    // [64,128]   }
  _Float16* nf     = qbuf;
  float*    scores = (float*)  (smem + 106496);    // [64,64] fp32
  _Float16* probs  = (_Float16*)(smem + 122880);   // [64,64]
  _Float16* avpre  = (_Float16*)(smem + 131072);   // [64,128], later h [64,64]
  _Float16* hbuf   = avpre;

  const int tid = threadIdx.x;
  const int b   = blockIdx.x;

  const _Float16* Wpre = ws + OFF_WPRE;
  const _Float16* Wk   = ws + OFF_WK;
  const _Float16* Wq   = ws + OFF_WQ;
  const _Float16* Wse  = ws + OFF_WSE;
  const _Float16* Wsa  = ws + OFF_WSA;
  const _Float16* Wav  = ws + OFF_WAV;
  const _Float16* Wsq  = ws + OFF_WSQ;
  const _Float16* Wf1  = ws + OFF_WF1;

  // 1) pack oa = [states | actions] as f16 (single HBM read of inputs)
  {
    const float* st = states  + (size_t)b * NAG * DOBS;
    const float* ac = actions + (size_t)b * NAG * NACT;
    for (int i = tid; i < NAG * DOA; i += 256) {
      const int n = i / DOA, c = i - n * DOA;
      const float v = (c < DOBS) ? st[n * DOBS + c] : ac[n * NACT + (c - DOBS)];
      oa[i] = (_Float16)v;
    }
  }
  __syncthreads();

  // 2) se_pre = lrelu(states @ Wpre^T + b_pre)  (also transposed copy)
  gemm_f16<DH, DOBS, true, true, true>(oa, DOA, Wpre, DOBS, sbufA, DH, sbufB, NAG, b_pre, tid);
  __syncthreads();

  // 3) q1, k1
  gemm_f16<DH, DH, false, false, false>(sbufA, DH, Wq, DH, qbuf, DH, nullptr, 0, nullptr, tid);
  gemm_f16<DH, DH, false, false, false>(sbufA, DH, Wk, DH, kbuf, DH, nullptr, 0, nullptr, tid);
  __syncthreads();

  // 4) scores1 = q1 @ k1^T * scale ; softmax -> probs (f16) + w1 (fp32, global)
  gemm_f32<DH>(qbuf, DH, kbuf, DH, scores, INV_SQRT_DK, tid);
  __syncthreads();
  softmax64(scores, probs, out_w1 + (size_t)b * NAG * NAG, tid);
  __syncthreads();

  // 5) av_pre = w1 @ se_pre   (B^T = se_preT)
  gemm_f16<DH, NAG, false, false, false>(probs, NAG, sbufB, NAG, avpre, DH, nullptr, 0, nullptr, tid);
  __syncthreads();

  // 6) se = lrelu(av_pre @ Wse^T + b_se)   (overwrites se_pre)
  gemm_f16<DH, DH, true, true, false>(avpre, DH, Wse, DH, sbufA, DH, nullptr, 0, b_se, tid);
  __syncthreads();

  // 7) q2, k2
  gemm_f16<DH, DH, false, false, false>(sbufA, DH, Wq, DH, qbuf, DH, nullptr, 0, nullptr, tid);
  gemm_f16<DH, DH, false, false, false>(sbufA, DH, Wk, DH, kbuf, DH, nullptr, 0, nullptr, tid);
  __syncthreads();

  // 8) scores2 ; softmax -> probs + w2 (global)
  gemm_f32<DH>(qbuf, DH, kbuf, DH, scores, INV_SQRT_DK, tid);
  __syncthreads();
  softmax64(scores, probs, out_w2 + (size_t)b * NAG * NAG, tid);
  __syncthreads();

  // 9) oae = lrelu(oa @ Wsa^T + b_sa)  -> sbufB region
  gemm_f16<DH, DOA, true, true, false>(oa, DOA, Wsa, DOA, sbufB, DH, nullptr, 0, b_sa, tid);
  __syncthreads();

  // 10) av = oae @ Wav^T (+ transposed copy); overwrites oa region (now free)
  gemm_f16<DH, DH, false, false, true>(sbufB, DH, Wav, DH, av, DH, avT, NAG, nullptr, tid);
  __syncthreads();

  // 11) nf = [ seq | agg ] : seq = lrelu(av_pre @ Wsq^T + b_sq), agg = w2 @ av
  gemm_f16<DH, DH, true, true, false>(avpre, DH, Wsq, DH, nf, 2 * DH, nullptr, 0, b_sq, tid);
  gemm_f16<DH, NAG, false, false, false>(probs, NAG, avT, NAG, nf + DH, 2 * DH, nullptr, 0, nullptr, tid);
  __syncthreads();

  // 12) h = lrelu(nf @ Wf1^T)
  gemm_f16<NAG, 2 * DH, true, false, false>(nf, 2 * DH, Wf1, 2 * DH, hbuf, NAG, nullptr, 0, nullptr, tid);
  __syncthreads();

  // 13) value = h @ Wf2^T
  if (tid < NAG) {
    float acc = 0.0f;
#pragma unroll
    for (int j = 0; j < 64; ++j) acc += (float)hbuf[tid * 64 + j] * W_f2[j];
    out_value[(size_t)b * NAG + tid] = acc;
  }
}

// ---------------- host launch ----------------
extern "C" void kernel_launch(void* const* d_in, const int* in_sizes, int n_in,
                              void* d_out, int out_size, void* d_ws, size_t ws_size,
                              hipStream_t stream) {
  const float* states  = (const float*)d_in[0];
  const float* actions = (const float*)d_in[1];
  const float* b_pre   = (const float*)d_in[3];
  const float* b_se    = (const float*)d_in[7];
  const float* b_sa    = (const float*)d_in[9];
  const float* b_sq    = (const float*)d_in[12];
  const float* W_f2    = (const float*)d_in[14];

  _Float16* ws = (_Float16*)d_ws;

  auto cvt = [&](int idx, int off, int n) {
    cvt_f32_f16<<<(n + 255) / 256, 256, 0, stream>>>((const float*)d_in[idx], ws + off, n);
  };
  cvt(2,  OFF_WPRE, DH * DOBS);    // W_pre
  cvt(4,  OFF_WK,   DH * DH);      // W_k
  cvt(5,  OFF_WQ,   DH * DH);      // W_q
  cvt(6,  OFF_WSE,  DH * DH);      // W_se
  cvt(8,  OFF_WSA,  DH * DOA);     // W_sa
  cvt(10, OFF_WAV,  DH * DH);      // W_av
  cvt(11, OFF_WSQ,  DH * DH);      // W_sq
  cvt(13, OFF_WF1,  64 * 2 * DH);  // W_f1

  float* out       = (float*)d_out;
  float* out_value = out;                                         // [B,N,1]
  float* out_w1    = out + (size_t)NB * NAG;                      // [B,N,N]
  float* out_w2    = out + (size_t)NB * NAG + (size_t)NB * NAG * NAG;

  (void)hipFuncSetAttribute((const void*)critic_fused,
                            hipFuncAttributeMaxDynamicSharedMemorySize, SMEM_BYTES);

  critic_fused<<<dim3(NB), dim3(256), SMEM_BYTES, stream>>>(
      states, actions, b_pre, b_se, b_sa, b_sq, W_f2, ws,
      out_value, out_w1, out_w2);
}
